// Block_1958505087147
// MI455X (gfx1250) — matrix operand order repaired and checked by reading, not compiled, not run on hardware
//
#include <hip/hip_runtime.h>
#include <stddef.h>
#include <stdint.h>

// ---------------- CDNA5 WMMA types & fragment helpers ----------------
typedef __attribute__((ext_vector_type(16))) __bf16 v16bf;
typedef __attribute__((ext_vector_type(8)))  float  v8f;

#define AS3 __attribute__((address_space(3)))

// LDS byte-offset of a generic pointer known to point into LDS
__device__ inline unsigned lds_off_of(const void* p) {
  return (unsigned)(uintptr_t)(AS3 const void*)p;
}

// async global->LDS 16-byte copy (per lane), tracked by ASYNCcnt
__device__ inline void async_ld_b128(unsigned lds_off, const void* gsrc) {
  asm volatile("global_load_async_to_lds_b128 %0, %1, off" ::"v"(lds_off),
               "v"((unsigned long long)(uintptr_t)gsrc)
               : "memory");
}
#define WAIT_ASYNC(n) asm volatile("s_wait_asynccnt " #n ::: "memory")

__device__ inline v8f wmma_bf16(v16bf a, v16bf b, v8f c) {
  return __builtin_amdgcn_wmma_f32_16x16x32_bf16(false, a, false, b, (short)0, c,
                                                 false, false);
}

__device__ inline v8f zero8() {
  v8f z = {0.f, 0.f, 0.f, 0.f, 0.f, 0.f, 0.f, 0.f};
  return z;
}

// A fragment (16x32 bf16, M x K), src row-major [m][k], stride ld.
// ISA 7.12.2: lanes 0-15 row M=lane, v0..3 = K0..7, v4..7 = K16..23;
// lanes 16-31 same rows, K8..15 / K24..31.
__device__ inline v16bf frag_a(const __bf16* src, int ld) {
  const int lane = threadIdx.x & 31;
  const __bf16* p = src + (lane & 15) * ld + ((lane >> 4) << 3);
  v16bf f;
#pragma unroll
  for (int e = 0; e < 8; ++e) f[e] = p[e];
#pragma unroll
  for (int e = 0; e < 8; ++e) f[8 + e] = p[16 + e];
  return f;
}

// B fragment (32x16 bf16, K x N), src row-major [k][n], stride ld.
// Column n = lane&15; lanes 0-15 hold K0..7,K16..23, lanes 16-31 hold K8..15,K24..31.
__device__ inline v16bf frag_b(const __bf16* src, int ld) {
  const int lane = threadIdx.x & 31;
  const __bf16* p = src + ((lane >> 4) << 3) * ld + (lane & 15);
  v16bf f;
#pragma unroll
  for (int e = 0; e < 8; ++e) f[e] = p[e * ld];
#pragma unroll
  for (int e = 0; e < 8; ++e) f[8 + e] = p[(16 + e) * ld];
  return f;
}

// B fragment where memory holds the transpose: src[n][k] row-major (e.g. K^T
// from K stored [key][d]).
__device__ inline v16bf frag_bt(const __bf16* src, int ld) {
  const int lane = threadIdx.x & 31;
  const __bf16* p = src + (lane & 15) * ld + ((lane >> 4) << 3);
  v16bf f;
#pragma unroll
  for (int e = 0; e < 8; ++e) f[e] = p[e];
#pragma unroll
  for (int e = 0; e < 8; ++e) f[8 + e] = p[16 + e];
  return f;
}

// ---------------- model constants ----------------
#define DIMC 768
#define HEADS 12
#define HDIM 64
#define HID2 2048
#define WS 14
#define NTOK 196     // 14*14
#define NPAD 208     // 13 * 16
#define NPAD2 224    // 7 * 32 (K padding for P@V)
#define NWIN 200     // 8 * 5 * 5
#define MROWS (NWIN * NPAD)   // 41600
#define MTOK 32768            // 8*64*64

// bf16 weight-block element offsets
#define WQKV_OFF 0u
#define WPROJ_OFF 1769472u           // 768*2304
#define W12_OFF  2359296u            // + 768*768
#define W3_OFF   5505024u            // + 768*4096
#define WB_ELEMS 7077888u            // + 2048*768
// bias (float) offsets within bias block
#define BQKV_OFF 0
#define BPROJ_OFF 2304
#define B12_OFF 3072
#define B3_OFF 7168
#define BIAS_ELEMS 7936

// ---------------- weight conversion / packing ----------------
__global__ __launch_bounds__(256) void cvt_weights(
    const float* __restrict__ qw, const float* __restrict__ kw,
    const float* __restrict__ vw, const float* __restrict__ pw,
    const float* __restrict__ w1, const float* __restrict__ w2,
    const float* __restrict__ w3, const float* __restrict__ qb,
    const float* __restrict__ vb, const float* __restrict__ pb,
    const float* __restrict__ b1, const float* __restrict__ b2,
    const float* __restrict__ b3, __bf16* __restrict__ WB,
    float* __restrict__ BF) {
  unsigned i = blockIdx.x * 256u + threadIdx.x;
  if (i < WPROJ_OFF) {  // QKV concat: [768][2304]
    unsigned k = i / 2304u, n = i % 2304u;
    float v = (n < 768u) ? qw[k * 768u + n]
              : (n < 1536u) ? kw[k * 768u + n - 768u]
                            : vw[k * 768u + n - 1536u];
    WB[i] = (__bf16)v;
  } else if (i < W12_OFF) {
    WB[i] = (__bf16)pw[i - WPROJ_OFF];
  } else if (i < W3_OFF) {  // interleave w1/w2 columns: [768][4096]
    unsigned j = i - W12_OFF;
    unsigned k = j / 4096u, n = j % 4096u;
    float v = (n & 1u) ? w2[k * 2048u + (n >> 1)] : w1[k * 2048u + (n >> 1)];
    WB[i] = (__bf16)v;
  } else if (i < WB_ELEMS) {
    WB[i] = (__bf16)w3[i - W3_OFF];
  } else if (i < WB_ELEMS + BIAS_ELEMS) {
    unsigned j = i - WB_ELEMS;
    float v;
    if (j < 2304u)
      v = (j < 768u) ? qb[j] : (j < 1536u) ? 0.f : vb[j - 1536u];
    else if (j < 3072u)
      v = pb[j - 2304u];
    else if (j < 7168u) {
      unsigned n = j - 3072u;
      v = (n & 1u) ? b2[n >> 1] : b1[n >> 1];
    } else
      v = b3[j - 7168u];
    BF[j] = v;
  }
}

// ---------------- block-wide 2-value reduction (blockDim == 256) -----------
__device__ inline void block_reduce2(float a, float b, float& A, float& B) {
  __shared__ float r1[8], r2[8];
  const int lane = threadIdx.x & 31, wave = threadIdx.x >> 5;
#pragma unroll
  for (int off = 16; off; off >>= 1) {
    a += __shfl_xor(a, off, 32);
    b += __shfl_xor(b, off, 32);
  }
  if (lane == 0) { r1[wave] = a; r2[wave] = b; }
  __syncthreads();
  if (threadIdx.x < 8) {
    a = r1[threadIdx.x];
    b = r2[threadIdx.x];
#pragma unroll
    for (int off = 4; off; off >>= 1) {
      a += __shfl_xor(a, off, 32);
      b += __shfl_xor(b, off, 32);
    }
    if (threadIdx.x == 0) { r1[0] = a; r2[0] = b; }
  }
  __syncthreads();
  A = r1[0];
  B = r2[0];
}

// ---------------- LN1 + window partition (zero-pad) -> bf16 t --------------
__global__ __launch_bounds__(256) void ln1_window(
    const float* __restrict__ x, const float* __restrict__ w,
    const float* __restrict__ b, __bf16* __restrict__ t) {
  const int row = blockIdx.x;  // 0..MROWS-1
  const int bw = row / NPAD, tok = row % NPAD;
  const int tid = threadIdx.x;
  bool valid = false;
  size_t src = 0;
  if (tok < NTOK) {
    int r14 = tok / WS, c14 = tok % WS;
    int bb = bw / 25, rem = bw % 25;
    int h = (rem / 5) * WS + r14, ww = (rem % 5) * WS + c14;
    if (h < 64 && ww < 64) {
      valid = true;
      src = (((size_t)bb * 64 + h) * 64 + ww) * DIMC;
    }
  }
  if (!valid) {
#pragma unroll
    for (int j = 0; j < 3; ++j)
      t[(size_t)row * DIMC + tid + j * 256] = (__bf16)0.f;
    return;
  }
  float v0 = x[src + tid], v1 = x[src + tid + 256], v2 = x[src + tid + 512];
  float S, S2;
  block_reduce2(v0 + v1 + v2, v0 * v0 + v1 * v1 + v2 * v2, S, S2);
  float mu = S * (1.f / DIMC);
  float rs = rsqrtf(S2 * (1.f / DIMC) - mu * mu + 1e-6f);
  float vv[3] = {v0, v1, v2};
#pragma unroll
  for (int j = 0; j < 3; ++j) {
    int c = tid + j * 256;
    t[(size_t)row * DIMC + c] = (__bf16)((vv[j] - mu) * rs * w[c] + b[c]);
  }
}

// ---------------- LN over rows of 768 fp32 -> bf16 -------------------------
__global__ __launch_bounds__(256) void ln_rows768(
    const float* __restrict__ in, const float* __restrict__ w,
    const float* __restrict__ b, __bf16* __restrict__ out) {
  const size_t row = blockIdx.x;
  const int tid = threadIdx.x;
  const float* p = in + row * DIMC;
  float v0 = p[tid], v1 = p[tid + 256], v2 = p[tid + 512];
  float S, S2;
  block_reduce2(v0 + v1 + v2, v0 * v0 + v1 * v1 + v2 * v2, S, S2);
  float mu = S * (1.f / DIMC);
  float rs = rsqrtf(S2 * (1.f / DIMC) - mu * mu + 1e-6f);
  float vv[3] = {v0, v1, v2};
#pragma unroll
  for (int j = 0; j < 3; ++j) {
    int c = tid + j * 256;
    out[row * DIMC + c] = (__bf16)((vv[j] - mu) * rs * w[c] + b[c]);
  }
}

// ---------------- LN over rows of 2048 bf16, in place ----------------------
__global__ __launch_bounds__(256) void ln_rows2048_inplace(
    __bf16* __restrict__ hid, const float* __restrict__ w,
    const float* __restrict__ b) {
  const size_t row = blockIdx.x;
  const int tid = threadIdx.x;
  float v[8];
  float s = 0.f, s2 = 0.f;
#pragma unroll
  for (int j = 0; j < 8; ++j) {
    v[j] = (float)hid[row * HID2 + tid + j * 256];
    s += v[j];
    s2 += v[j] * v[j];
  }
  float S, S2;
  block_reduce2(s, s2, S, S2);
  float mu = S * (1.f / HID2);
  float rs = rsqrtf(S2 * (1.f / HID2) - mu * mu + 1e-6f);
#pragma unroll
  for (int j = 0; j < 8; ++j) {
    int c = tid + j * 256;
    hid[row * HID2 + c] = (__bf16)((v[j] - mu) * rs * w[c] + b[c]);
  }
}

// ---------------- generic 128x128 WMMA GEMM with fused epilogues -----------
enum { EPI_BF16 = 0, EPI_PROJ = 1, EPI_SWIGLU = 2, EPI_W3RES = 3 };

// issue one 128x32 A-tile + 32x128 B-tile stage as async global->LDS copies
__device__ inline void gemm_async_issue(const __bf16* A, const __bf16* B,
                                        __bf16* lA, __bf16* lB, int bm, int bn,
                                        int k0, int K, int N) {
  const int tid = threadIdx.x;
#pragma unroll
  for (int j = 0; j < 2; ++j) {
    const int e8 = (tid + j * 256) * 8;  // 8 bf16 = 16 bytes per lane
    const int r = e8 >> 5, c = e8 & 31;
    async_ld_b128(lds_off_of(&lA[e8]), &A[(size_t)(bm + r) * K + k0 + c]);
    const int r2 = e8 >> 7, c2 = e8 & 127;
    async_ld_b128(lds_off_of(&lB[e8]), &B[(size_t)(k0 + r2) * N + bn + c2]);
  }
}

template <int EPI>
__global__ __launch_bounds__(256) void gemm_wmma(
    const __bf16* __restrict__ A, const __bf16* __restrict__ B,
    const float* __restrict__ bias, __bf16* __restrict__ outb,
    float* __restrict__ outf, const float* __restrict__ xres, int M, int N,
    int K) {
  const int tid = threadIdx.x;
  const int lane = tid & 31, wave = tid >> 5;
  const int wm = (wave >> 2) * 64, wn = (wave & 3) * 32;
  const int bm = (int)blockIdx.x * 128;
  const int bn = (int)blockIdx.y * 128;
  v8f acc[4][2];
#pragma unroll
  for (int mt = 0; mt < 4; ++mt)
#pragma unroll
    for (int nt = 0; nt < 2; ++nt) acc[mt][nt] = zero8();

  __shared__ __bf16 As[2][128 * 32];
  __shared__ __bf16 Bs[2][32 * 128];
  const int nk = K >> 5;
  // double-buffered async pipeline: stage kk+1 copies overlap stage kk WMMAs
  gemm_async_issue(A, B, As[0], Bs[0], bm, bn, 0, K, N);
  for (int kk = 0; kk < nk; ++kk) {
    const int cur = kk & 1;
    if (kk + 1 < nk) {
      gemm_async_issue(A, B, As[cur ^ 1], Bs[cur ^ 1], bm, bn, (kk + 1) << 5,
                       K, N);
      WAIT_ASYNC(4);  // 4 issues/stage per wave, in-order => stage kk landed
    } else {
      WAIT_ASYNC(0);
    }
    __syncthreads();
    v16bf af[4], bfr[2];
#pragma unroll
    for (int mt = 0; mt < 4; ++mt)
      af[mt] = frag_a(&As[cur][(wm + mt * 16) * 32], 32);
#pragma unroll
    for (int nt = 0; nt < 2; ++nt) bfr[nt] = frag_b(&Bs[cur][wn + nt * 16], 128);
#pragma unroll
    for (int mt = 0; mt < 4; ++mt)
#pragma unroll
      for (int nt = 0; nt < 2; ++nt)
        acc[mt][nt] = wmma_bf16(af[mt], bfr[nt], acc[mt][nt]);
    __syncthreads();  // protect buffer reuse at stage kk+2
  }

  const int hf = lane >> 4, nl = lane & 15;
#pragma unroll
  for (int mt = 0; mt < 4; ++mt) {
#pragma unroll
    for (int nt = 0; nt < 2; ++nt) {
      const int n = bn + wn + nt * 16 + nl;
#pragma unroll
      for (int r = 0; r < 8; ++r) {
        const int m = bm + wm + mt * 16 + r + hf * 8;
        float v = acc[mt][nt][r];
        if (EPI == EPI_BF16) {
          outb[(size_t)m * N + n] = (__bf16)(v + bias[n]);
        } else if (EPI == EPI_W3RES) {
          outf[(size_t)m * N + n] += v + bias[n];
        } else if (EPI == EPI_PROJ) {
          // window token -> image pixel scatter + residual add of x
          int bw = m / NPAD, tok = m % NPAD;
          if (tok < NTOK) {
            int r14 = tok / WS, c14 = tok % WS;
            int bb = bw / 25, rem = bw % 25;
            int h = (rem / 5) * WS + r14, ww = (rem % 5) * WS + c14;
            if (h < 64 && ww < 64) {
              size_t idx = (((size_t)bb * 64 + h) * 64 + ww) * DIMC + n;
              outf[idx] = xres[idx] + v + bias[n];
            }
          }
        } else {  // EPI_SWIGLU: even col = w1 path, odd col = w2 path
          float other = __shfl_xor(v, 1, 32);
          if ((n & 1) == 0) {
            float a1 = v + bias[n];
            float a2 = other + bias[n + 1];
            float sg = a1 / (1.f + __expf(-a1));
            outb[(size_t)m * (N >> 1) + (n >> 1)] = (__bf16)(sg * a2);
          }
        }
      }
    }
  }
}

// ---------------- RoPE angle ----------------
__device__ inline void rope_cs(int tok, int d, float* c, float* s) {
  int p = (d < 32) ? (tok / WS) : (tok % WS);
  int i = (d & 31) >> 1;
  // freq = 10000^(-i/16) = exp2(-i * log2(10000)/16); v_exp_f32 path
  float freq = __builtin_amdgcn_exp2f(-(float)i * 0.83048202372184f);
  float ang = (float)p * (16.0f / 14.0f) * freq;
  *c = __cosf(ang);
  *s = __sinf(ang);
}

// ---------------- window attention: one block per (window, head) -----------
__global__ __launch_bounds__(128) void attn_kernel(
    const __bf16* __restrict__ qkv, __bf16* __restrict__ oatt) {
  extern __shared__ __bf16 smem[];
  __bf16* Ks = smem;                        // [208][64]
  __bf16* Vs = Ks + NPAD * HDIM;            // [224][64]
  __bf16* Pball = Vs + NPAD2 * HDIM;        // [4][16][224]
  const int bw = blockIdx.x / HEADS, head = blockIdx.x % HEADS;
  const int tid = threadIdx.x, lane = tid & 31, wave = tid >> 5;
  __bf16* Pb = Pball + wave * (16 * NPAD2);
  const size_t rowbase = (size_t)bw * NPAD;

  // V -> LDS via async copy (pure data movement, no transform needed)
  for (int cch = tid; cch < NTOK * 8; cch += 128) {
    const int tok = cch >> 3, col = (cch & 7) * 8;  // 16B chunks
    async_ld_b128(lds_off_of(&Vs[tok * HDIM + col]),
                  &qkv[(rowbase + tok) * 2304 + 1536 + head * HDIM + col]);
  }
  for (int i = tid; i < (NPAD2 - NTOK) * (HDIM / 2); i += 128)
    *(unsigned*)&Vs[NTOK * HDIM + i * 2] = 0u;

  // K -> LDS with RoPE (pairs), zero beyond 196 tokens
  for (int i = tid; i < NPAD * 32; i += 128) {
    int tok = i >> 5, d = (i & 31) * 2;
    __bf16 o0 = (__bf16)0.f, o1 = (__bf16)0.f;
    if (tok < NTOK) {
      const __bf16* src = &qkv[(rowbase + tok) * 2304 + 768 + head * HDIM + d];
      float k0 = (float)src[0], k1 = (float)src[1];
      float cs, sn;
      rope_cs(tok, d, &cs, &sn);
      o0 = (__bf16)(k0 * cs - k1 * sn);
      o1 = (__bf16)(k1 * cs + k0 * sn);
    }
    Ks[tok * HDIM + d] = o0;
    Ks[tok * HDIM + d + 1] = o1;
  }
  // zero this wave's P pad columns 208..223 once
  for (int i = lane; i < 16 * 16; i += 32)
    Pb[(i >> 4) * NPAD2 + NPAD + (i & 15)] = (__bf16)0.f;
  WAIT_ASYNC(0);  // V tiles landed
  __syncthreads();

  for (int rt = wave; rt < 13; rt += 4) {
    // Q fragments (2 k-steps of 32 over HDIM=64) with RoPE + 1/sqrt(64)
    v16bf aq[2];
    {
      const int mrow = lane & 15, hf = lane >> 4;
      const int tok = rt * 16 + mrow;
#pragma unroll
      for (int ks = 0; ks < 2; ++ks) {
        v16bf f;
        if (tok < NTOK) {
          const __bf16* src =
              &qkv[(rowbase + tok) * 2304 + head * HDIM + ks * 32 + hf * 8];
#pragma unroll
          for (int e = 0; e < 8; e += 2) {
            int d = ks * 32 + hf * 8 + e;
            float q0 = (float)src[e], q1 = (float)src[e + 1];
            float cs, sn;
            rope_cs(tok, d, &cs, &sn);
            f[e] = (__bf16)(0.125f * (q0 * cs - q1 * sn));
            f[e + 1] = (__bf16)(0.125f * (q1 * cs + q0 * sn));
          }
#pragma unroll
          for (int e = 0; e < 8; e += 2) {
            int d = ks * 32 + hf * 8 + 16 + e;
            float q0 = (float)src[16 + e], q1 = (float)src[16 + e + 1];
            float cs, sn;
            rope_cs(tok, d, &cs, &sn);
            f[8 + e] = (__bf16)(0.125f * (q0 * cs - q1 * sn));
            f[8 + e + 1] = (__bf16)(0.125f * (q1 * cs + q0 * sn));
          }
        } else {
#pragma unroll
          for (int e = 0; e < 16; ++e) f[e] = (__bf16)0.f;
        }
        aq[ks] = f;
      }
    }
    // scores S = Q Kt : 13 column tiles
    v8f sc[13];
#pragma unroll
    for (int ct = 0; ct < 13; ++ct) sc[ct] = zero8();
#pragma unroll
    for (int ct = 0; ct < 13; ++ct)
#pragma unroll
      for (int ks = 0; ks < 2; ++ks)
        sc[ct] = wmma_bf16(aq[ks], frag_bt(&Ks[(ct * 16) * HDIM + ks * 32], HDIM),
                           sc[ct]);
    // mask key columns 196..207
    if ((lane & 15) >= 4) {
#pragma unroll
      for (int r = 0; r < 8; ++r) sc[12][r] = -1e30f;
    }
    // softmax (rows split across the two 16-lane halves)
    float mx[8], sum[8];
#pragma unroll
    for (int r = 0; r < 8; ++r) {
      float m = sc[0][r];
#pragma unroll
      for (int ct = 1; ct < 13; ++ct) m = fmaxf(m, sc[ct][r]);
#pragma unroll
      for (int off = 8; off; off >>= 1) m = fmaxf(m, __shfl_xor(m, off, 32));
      mx[r] = m;
    }
#pragma unroll
    for (int r = 0; r < 8; ++r) sum[r] = 0.f;
#pragma unroll
    for (int ct = 0; ct < 13; ++ct)
#pragma unroll
      for (int r = 0; r < 8; ++r) {
        float e = __expf(sc[ct][r] - mx[r]);
        sc[ct][r] = e;
        sum[r] += e;
      }
#pragma unroll
    for (int r = 0; r < 8; ++r) {
#pragma unroll
      for (int off = 8; off; off >>= 1) sum[r] += __shfl_xor(sum[r], off, 32);
      sum[r] = 1.f / sum[r];
    }
    // write normalized P (bf16) to this wave's LDS tile [16][224]
    {
      const int hf = lane >> 4, nl = lane & 15;
#pragma unroll
      for (int ct = 0; ct < 13; ++ct)
#pragma unroll
        for (int r = 0; r < 8; ++r)
          Pb[(r + hf * 8) * NPAD2 + ct * 16 + nl] =
              (__bf16)(sc[ct][r] * sum[r]);
    }
    asm volatile("s_wait_dscnt 0x0" ::: "memory");
    // O = P V : 7 k-steps of 32 over padded 224, 4 output column tiles
    v8f ov[4];
#pragma unroll
    for (int nt = 0; nt < 4; ++nt) ov[nt] = zero8();
#pragma unroll
    for (int ks2 = 0; ks2 < 7; ++ks2) {
      v16bf ap = frag_a(&Pb[ks2 * 32], NPAD2);
#pragma unroll
      for (int nt = 0; nt < 4; ++nt)
        ov[nt] = wmma_bf16(ap, frag_b(&Vs[(ks2 * 32) * HDIM + nt * 16], HDIM),
                           ov[nt]);
    }
    {
      const int hf = lane >> 4, nl = lane & 15;
#pragma unroll
      for (int nt = 0; nt < 4; ++nt)
#pragma unroll
        for (int r = 0; r < 8; ++r)
          oatt[(rowbase + rt * 16 + r + hf * 8) * DIMC + head * HDIM +
               nt * 16 + nl] = (__bf16)ov[nt][r];
    }
    asm volatile("s_wait_dscnt 0x0" ::: "memory");
  }
}

// ---------------- host-side orchestration ----------------
extern "C" void kernel_launch(void* const* d_in, const int* in_sizes, int n_in,
                              void* d_out, int out_size, void* d_ws,
                              size_t ws_size, hipStream_t stream) {
  const float* x = (const float*)d_in[0];
  const float* ln1_w = (const float*)d_in[1];
  const float* ln1_b = (const float*)d_in[2];
  const float* q_w = (const float*)d_in[3];
  const float* q_b = (const float*)d_in[4];
  const float* k_w = (const float*)d_in[5];
  const float* v_w = (const float*)d_in[6];
  const float* v_b = (const float*)d_in[7];
  const float* proj_w = (const float*)d_in[8];
  const float* proj_b = (const float*)d_in[9];
  const float* ln2_w = (const float*)d_in[10];
  const float* ln2_b = (const float*)d_in[11];
  const float* w1_w = (const float*)d_in[12];
  const float* w1_b = (const float*)d_in[13];
  const float* w2_w = (const float*)d_in[14];
  const float* w2_b = (const float*)d_in[15];
  const float* ffn_w = (const float*)d_in[16];
  const float* ffn_b = (const float*)d_in[17];
  const float* w3_w = (const float*)d_in[18];
  const float* w3_b = (const float*)d_in[19];
  float* out = (float*)d_out;

  char* wsb = (char*)d_ws;
  __bf16* WB = (__bf16*)wsb;                          // 16 MB weight block
  float* BF = (float*)(wsb + (size_t)WB_ELEMS * 2);   // biases
  __bf16* T = (__bf16*)(wsb + ((size_t)16 << 20));    // t, later o_attn
  __bf16* QKV = (__bf16*)(wsb + ((size_t)80 << 20));  // qkv, later h2
  __bf16* HID = (__bf16*)(wsb + ((size_t)272 << 20)); // swiglu output

  const __bf16* Wqkv = WB + WQKV_OFF;
  const __bf16* Wproj = WB + WPROJ_OFF;
  const __bf16* W12 = WB + W12_OFF;
  const __bf16* W3 = WB + W3_OFF;
  const float* Bqkv = BF + BQKV_OFF;
  const float* Bproj = BF + BPROJ_OFF;
  const float* B12 = BF + B12_OFF;
  const float* B3 = BF + B3_OFF;

  // 1) pack weights/biases to bf16 (QKV concat, w1/w2 interleaved)
  cvt_weights<<<(WB_ELEMS + BIAS_ELEMS) / 256, 256, 0, stream>>>(
      q_w, k_w, v_w, proj_w, w1_w, w2_w, w3_w, q_b, v_b, proj_b, w1_b, w2_b,
      w3_b, WB, BF);
  // 2) LN1 + window partition -> t (bf16, zero-padded to 200x208 tokens)
  ln1_window<<<MROWS, 256, 0, stream>>>(x, ln1_w, ln1_b, T);
  // 3) fused QKV GEMM: [41600,768] x [768,2304]
  {
    dim3 g(MROWS / 128, 2304 / 128);
    gemm_wmma<EPI_BF16><<<g, 256, 0, stream>>>(T, Wqkv, Bqkv, QKV, nullptr,
                                               nullptr, MROWS, 2304, DIMC);
  }
  // 4) windowed attention with fused RoPE; writes o_attn over T
  {
    size_t smem = (size_t)(NPAD * HDIM + NPAD2 * HDIM + 4 * 16 * NPAD2) * 2;
    attn_kernel<<<NWIN * HEADS, 128, smem, stream>>>(QKV, T);
  }
  // 5) proj GEMM with fused window->image scatter + residual into d_out
  {
    dim3 g(MROWS / 128, DIMC / 128);
    gemm_wmma<EPI_PROJ><<<g, 256, 0, stream>>>(T, Wproj, Bproj, nullptr, out,
                                               x, MROWS, DIMC, DIMC);
  }
  // 6) LN2 on x1 (d_out) -> h2 (reuse QKV buffer)
  ln_rows768<<<MTOK, 256, 0, stream>>>(out, ln2_w, ln2_b, QKV);
  // 7) fused SwiGLU GEMM: interleaved w1/w2, epilogue silu(a)*b -> HID
  {
    dim3 g(MTOK / 128, 4096 / 128);
    gemm_wmma<EPI_SWIGLU><<<g, 256, 0, stream>>>(QKV, W12, B12, HID, nullptr,
                                                 nullptr, MTOK, 4096, DIMC);
  }
  // 8) FFN LayerNorm over 2048, in place
  ln_rows2048_inplace<<<MTOK, 256, 0, stream>>>(HID, ffn_w, ffn_b);
  // 9) w3 GEMM, accumulate into d_out (residual)
  {
    dim3 g(MTOK / 128, DIMC / 128);
    gemm_wmma<EPI_W3RES><<<g, 256, 0, stream>>>(HID, W3, B3, nullptr, out,
                                                nullptr, MTOK, DIMC, HID2);
  }
  (void)in_sizes; (void)n_in; (void)out_size; (void)ws_size;
}
